// Attention_81406810128551
// MI455X (gfx1250) — matrix operand order repaired and checked by reading, not compiled, not run on hardware
//
#include <hip/hip_runtime.h>
#include <hip/hip_bf16.h>

typedef __attribute__((ext_vector_type(16))) __bf16 v16bf;
typedef __attribute__((ext_vector_type(8)))  __bf16 v8bf;
typedef __attribute__((ext_vector_type(8)))  float  v8f;

union V16b { v16bf v; v8bf h[2]; };

__device__ inline unsigned short f2bf(float f) {
    union { float f; unsigned int u; } c; c.f = f;
    unsigned int u = c.u + 0x7FFFu + ((c.u >> 16) & 1u);   // RNE
    return (unsigned short)(u >> 16);
}
__device__ inline float bf2f(unsigned short s) {
    union { unsigned int u; float f; } c; c.u = ((unsigned int)s) << 16;
    return c.f;
}

// ---------------- pack kernels ----------------
__global__ void pack_x_kernel(const float* __restrict__ x,
                              unsigned short* __restrict__ out, int n) {
    int i = blockIdx.x * blockDim.x + threadIdx.x;
    if (i < n) out[i] = f2bf(x[i]);
}

// Wt[n][k] = bf16(W[k][n]) so B-operand loads are contiguous in k.
__global__ void pack_wT_kernel(const float* __restrict__ W,
                               unsigned short* __restrict__ out) {
    int i = blockIdx.x * blockDim.x + threadIdx.x;
    if (i < 768 * 768) {
        int n = i / 768, k = i % 768;
        out[i] = f2bf(W[k * 768 + n]);
    }
}

// ---------------- generic bf16 GEMM: C(4096x768) = A(4096x768) @ Wt^T + bias --
// One wave computes a 64x32 output tile (4x2 WMMA accumulators) to raise
// arithmetic intensity to ~22 FLOP/B from L2.
// mode 0: q  -> bf16 (48,1024,64)
// mode 1: k  -> bf16 (48,1024,64), pre-scaled by 1/8 (exact power of two)
// mode 2: v  -> bf16 transposed (48,64,1024)
// mode 3: out-> f32 (4096,768)
__global__ __launch_bounds__(256) void gemm_bf16_kernel(
    const unsigned short* __restrict__ A,
    const unsigned short* __restrict__ Bt,
    const float* __restrict__ bias, int mode,
    unsigned short* __restrict__ out_q,
    unsigned short* __restrict__ out_k,
    unsigned short* __restrict__ out_vT,
    float* __restrict__ out_f)
{
    const int K = 768, NT = 24;                  // 24 n-tiles of 32
    int wave = (blockIdx.x * blockDim.x + threadIdx.x) >> 5;
    int lane = threadIdx.x & 31;
    int hh = lane >> 4, ln = lane & 15;
    int m0 = (wave / NT) * 64;
    int n0 = (wave % NT) * 32;

    v8f acc[4][2] = {};
    for (int kk = 0; kk < K; kk += 32) {
        V16b a[4], b[2];
#pragma unroll
        for (int mi = 0; mi < 4; ++mi) {
            const unsigned short* ap = A + (size_t)(m0 + mi * 16 + ln) * K + kk;
            a[mi].h[0] = *(const v8bf*)(ap + hh * 8);
            a[mi].h[1] = *(const v8bf*)(ap + 16 + hh * 8);
        }
#pragma unroll
        for (int ni = 0; ni < 2; ++ni) {
            const unsigned short* bp = Bt + (size_t)(n0 + ni * 16 + ln) * K + kk + hh * 8;
            b[ni].h[0] = *(const v8bf*)(bp);
            b[ni].h[1] = *(const v8bf*)(bp + 16);
        }
#pragma unroll
        for (int mi = 0; mi < 4; ++mi)
#pragma unroll
            for (int ni = 0; ni < 2; ++ni)
                acc[mi][ni] = __builtin_amdgcn_wmma_f32_16x16x32_bf16(
                    false, a[mi].v, false, b[ni].v, (short)0, acc[mi][ni], false, false);
    }
#pragma unroll
    for (int mi = 0; mi < 4; ++mi)
#pragma unroll
        for (int ni = 0; ni < 2; ++ni)
#pragma unroll
            for (int e = 0; e < 8; ++e) {
                int row = m0 + mi * 16 + e + 8 * hh;
                int col = n0 + ni * 16 + ln;
                float val = acc[mi][ni][e] + bias[col];
                int bidx = row >> 10, t = row & 1023;
                int nh = col >> 6, d = col & 63;
                int bh = bidx * 12 + nh;
                if (mode == 0)
                    out_q[((size_t)bh * 1024 + t) * 64 + d] = f2bf(val);
                else if (mode == 1)
                    out_k[((size_t)bh * 1024 + t) * 64 + d] = f2bf(val * 0.125f);
                else if (mode == 2)
                    out_vT[((size_t)bh * 64 + d) * 1024 + t] = f2bf(val);
                else
                    out_f[(size_t)row * 768 + col] = val;
            }
}

// ---------------- fused attention (flash-style, 1 wave per (bh, 16-row qtile))
__global__ __launch_bounds__(32) void attn_kernel(
    const unsigned short* __restrict__ qbf,  // (48,1024,64)
    const unsigned short* __restrict__ kbf,  // (48,1024,64), pre-scaled
    const unsigned short* __restrict__ vT,   // (48,64,1024)
    const float* __restrict__ rel_h,         // (63,64)
    const float* __restrict__ rel_w,         // (63,64)
    unsigned short* __restrict__ obf)        // (4096,768) bf16
{
    __shared__ float qs[16 * 64];
    __shared__ float bh_lds[16 * 32];
    __shared__ float bw_lds[16 * 32];
    __shared__ __align__(16) unsigned short p_lds[16 * 32];

    int bh = blockIdx.x / 64;
    int qt = blockIdx.x % 64;
    int lane = threadIdx.x;
    int hh = lane >> 4, ln = lane & 15;

    size_t qrowbase = ((size_t)bh * 1024 + qt * 16) * 64;
    for (int i = lane; i < 16 * 64; i += 32) qs[i] = bf2f(qbf[qrowbase + i]);
    __syncthreads();

    // decomposed rel-pos bias: 16 rows x 32 (h) + 16 x 32 (w), f32 dots of len 64
    for (int idx = lane; idx < 1024; idx += 32) {
        int which = idx >> 9;
        int r = (idx >> 5) & 15;
        int j = idx & 31;
        int t = qt * 16 + r;
        const float* tab = which ? (rel_w + (((t & 31) - j + 31) * 64))
                                 : (rel_h + (((t >> 5) - j + 31) * 64));
        const float* qr = qs + r * 64;
        float s = 0.f;
        for (int d = 0; d < 64; ++d) s += qr[d] * tab[d];
        if (which) bw_lds[r * 32 + j] = s; else bh_lds[r * 32 + j] = s;
    }
    __syncthreads();

    // q A-operands for both K-steps (HD=64 -> 2x K=32)
    V16b qa[2];
    {
        const unsigned short* ap = qbf + qrowbase + (size_t)ln * 64;
#pragma unroll
        for (int ks = 0; ks < 2; ++ks) {
            qa[ks].h[0] = *(const v8bf*)(ap + ks * 32 + hh * 8);
            qa[ks].h[1] = *(const v8bf*)(ap + ks * 32 + 16 + hh * 8);
        }
    }

    float mrun[8], lrun[8];
#pragma unroll
    for (int e = 0; e < 8; ++e) { mrun[e] = -1e30f; lrun[e] = 0.f; }
    v8f o[4] = {};

    for (int kc = 0; kc < 32; ++kc) {
        int keybase = kc * 32;
        v8f s[2];
        // seed score accumulators with the rel-pos bias (WMMA C operand)
#pragma unroll
        for (int half = 0; half < 2; ++half)
#pragma unroll
            for (int e = 0; e < 8; ++e) {
                int m = e + 8 * hh;
                int key = keybase + half * 16 + ln;
                s[half][e] = bh_lds[m * 32 + (key >> 5)] + bw_lds[m * 32 + (key & 31)];
            }
        // S = q @ k^T / 8 + bias
#pragma unroll
        for (int half = 0; half < 2; ++half) {
            int key = keybase + half * 16 + ln;
            const unsigned short* kp = kbf + ((size_t)bh * 1024 + key) * 64;
#pragma unroll
            for (int ks = 0; ks < 2; ++ks) {
                V16b b;
                b.h[0] = *(const v8bf*)(kp + ks * 32 + hh * 8);
                b.h[1] = *(const v8bf*)(kp + ks * 32 + 16 + hh * 8);
                s[half] = __builtin_amdgcn_wmma_f32_16x16x32_bf16(
                    false, qa[ks].v, false, b.v, (short)0, s[half], false, false);
            }
        }
        // online softmax over the 32-key chunk (rows live in 16-lane groups)
        float p0[8], p1[8];
#pragma unroll
        for (int e = 0; e < 8; ++e) {
            float v = fmaxf(s[0][e], s[1][e]);
#pragma unroll
            for (int msk = 1; msk < 16; msk <<= 1) v = fmaxf(v, __shfl_xor(v, msk, 16));
            float mnew = fmaxf(mrun[e], v);
            float sc = __expf(mrun[e] - mnew);
            p0[e] = __expf(s[0][e] - mnew);
            p1[e] = __expf(s[1][e] - mnew);
            float rs = p0[e] + p1[e];
#pragma unroll
            for (int msk = 1; msk < 16; msk <<= 1) rs += __shfl_xor(rs, msk, 16);
            lrun[e] = lrun[e] * sc + rs;
            mrun[e] = mnew;
#pragma unroll
            for (int ni = 0; ni < 4; ++ni) o[ni][e] *= sc;
        }
        // stage P (16x32) to LDS, reload in A-matrix layout
#pragma unroll
        for (int e = 0; e < 8; ++e) {
            int m = e + 8 * hh;
            p_lds[m * 32 + ln]      = f2bf(p0[e]);
            p_lds[m * 32 + 16 + ln] = f2bf(p1[e]);
        }
        __syncthreads();
        V16b pa;
        pa.h[0] = *(const v8bf*)(p_lds + ln * 32 + hh * 8);
        pa.h[1] = *(const v8bf*)(p_lds + ln * 32 + 16 + hh * 8);
        // O += P @ V
#pragma unroll
        for (int ni = 0; ni < 4; ++ni) {
            int n = ni * 16 + ln;
            const unsigned short* vp = vT + ((size_t)bh * 64 + n) * 1024 + keybase + hh * 8;
            V16b b;
            b.h[0] = *(const v8bf*)(vp);
            b.h[1] = *(const v8bf*)(vp + 16);
            o[ni] = __builtin_amdgcn_wmma_f32_16x16x32_bf16(
                false, pa.v, false, b.v, (short)0, o[ni], false, false);
        }
        __syncthreads();
    }

    // normalize and scatter to (b, token, nh*64+d) bf16 for the output projection
    int b = bh / 12, nh = bh % 12;
#pragma unroll
    for (int e = 0; e < 8; ++e) {
        float inv = 1.0f / lrun[e];
        int t = qt * 16 + e + 8 * hh;
        size_t gr = (size_t)b * 1024 + t;
#pragma unroll
        for (int ni = 0; ni < 4; ++ni) {
            int col = nh * 64 + ni * 16 + ln;
            obf[gr * 768 + col] = f2bf(o[ni][e] * inv);
        }
    }
}

extern "C" void kernel_launch(void* const* d_in, const int* in_sizes, int n_in,
                              void* d_out, int out_size, void* d_ws, size_t ws_size,
                              hipStream_t stream) {
    const float* hs    = (const float*)d_in[0];
    const float* Wq    = (const float*)d_in[1];
    const float* bq    = (const float*)d_in[2];
    const float* Wk    = (const float*)d_in[3];
    const float* bk    = (const float*)d_in[4];
    const float* Wv    = (const float*)d_in[5];
    const float* bv    = (const float*)d_in[6];
    const float* Wp    = (const float*)d_in[7];
    const float* bp    = (const float*)d_in[8];
    const float* rel_h = (const float*)d_in[9];
    const float* rel_w = (const float*)d_in[10];

    size_t off = 0;
    auto carve = [&](size_t bytes) {
        void* p = (char*)d_ws + off;
        off += (bytes + 255) & ~(size_t)255;
        return p;
    };
    unsigned short* x_bf   = (unsigned short*)carve((size_t)4096 * 768 * 2);
    unsigned short* wq_t   = (unsigned short*)carve((size_t)768 * 768 * 2);
    unsigned short* wk_t   = (unsigned short*)carve((size_t)768 * 768 * 2);
    unsigned short* wv_t   = (unsigned short*)carve((size_t)768 * 768 * 2);
    unsigned short* wp_t   = (unsigned short*)carve((size_t)768 * 768 * 2);
    unsigned short* q_bf   = (unsigned short*)carve((size_t)48 * 1024 * 64 * 2);
    unsigned short* k_bf   = (unsigned short*)carve((size_t)48 * 1024 * 64 * 2);
    unsigned short* vT_bf  = (unsigned short*)carve((size_t)48 * 64 * 1024 * 2);
    unsigned short* att_bf = (unsigned short*)carve((size_t)4096 * 768 * 2);

    const int NX = 4096 * 768;
    pack_x_kernel<<<(NX + 255) / 256, 256, 0, stream>>>(hs, x_bf, NX);
    pack_wT_kernel<<<(768 * 768) / 256, 256, 0, stream>>>(Wq, wq_t);
    pack_wT_kernel<<<(768 * 768) / 256, 256, 0, stream>>>(Wk, wk_t);
    pack_wT_kernel<<<(768 * 768) / 256, 256, 0, stream>>>(Wv, wv_t);
    pack_wT_kernel<<<(768 * 768) / 256, 256, 0, stream>>>(Wp, wp_t);

    // 1536 wave-tiles (64 x 24) per GEMM, 8 waves/block -> 192 blocks
    gemm_bf16_kernel<<<192, 256, 0, stream>>>(x_bf, wq_t, bq, 0, q_bf, k_bf, vT_bf, nullptr);
    gemm_bf16_kernel<<<192, 256, 0, stream>>>(x_bf, wk_t, bk, 1, q_bf, k_bf, vT_bf, nullptr);
    gemm_bf16_kernel<<<192, 256, 0, stream>>>(x_bf, wv_t, bv, 2, q_bf, k_bf, vT_bf, nullptr);

    attn_kernel<<<48 * 64, 32, 0, stream>>>(q_bf, k_bf, vT_bf, rel_h, rel_w, att_bf);

    gemm_bf16_kernel<<<192, 256, 0, stream>>>(att_bf, wp_t, bp, 3, q_bf, k_bf, vT_bf,
                                              (float*)d_out);
}
